// Attention_81870666597078
// MI455X (gfx1250) — matrix operand order repaired and checked by reading, not compiled, not run on hardware
//
#include <hip/hip_runtime.h>

// ---------------------------------------------------------------------------
// MHA forward for MI455X (gfx1250, wave32, WMMA + TDM + async-LDS).
//   B=4, S=2048, D=1024, H=16, DH=64.
// Pipeline: cast->f16, QKV GEMM (WMMA, TDM double-buffered), flash-attention
// (WMMA, async global->LDS), out GEMM (WMMA, TDM).
// Compute-bound (~140 GFLOP vs ~6us of HBM traffic at 23.3 TB/s), so all
// matrix math runs through v_wmma_f32_16x16x32_f16 with f32 accumulation, and
// tile staging is offloaded to the Tensor Data Mover / async-LDS engines so
// WMMA issue slots aren't spent on copies.
// ---------------------------------------------------------------------------

typedef __attribute__((ext_vector_type(16))) _Float16 v16h;
typedef __attribute__((ext_vector_type(8)))  float    v8f;
typedef __attribute__((ext_vector_type(8)))  _Float16 h8;
typedef __attribute__((ext_vector_type(4)))  unsigned su4;
typedef __attribute__((ext_vector_type(8)))  unsigned su8;

union FragU { v16h v; unsigned u[8]; };

// CDNA5 16-bit A-matrix 16x32 layout (ISA 7.12.2):
//   lanes 0-15 : M=lane;    vgpr v holds K pair at (v/4)*16 + (v%4)*2
//   lanes 16-31: M=lane-16; same + 8
__device__ inline v16h load_fragA(const _Float16* base, int stride) {
  const int l  = threadIdx.x & 31;
  const int hi = l >> 4;
  const _Float16* p = base + (size_t)(l & 15) * stride;
  FragU f;
#pragma unroll
  for (int v = 0; v < 8; ++v) {
    const int k = ((v >> 2) << 4) | (hi << 3) | ((v & 3) << 1);
    f.u[v] = *(const unsigned*)(p + k);
  }
  return f.v;
}

// CDNA5 16-bit B-matrix 32x16 layout (same K order as A):
//   lanes 0-15 : N=lane;    vgpr v holds K pair at (v/4)*16 + (v%4)*2
//   lanes 16-31: N=lane-16; same + 8
__device__ inline v16h load_fragB(const _Float16* base, int stride) {
  const int l  = threadIdx.x & 31;
  const int hi = l >> 4;
  const _Float16* p = base + (size_t)(l & 15) * stride;
  FragU f;
#pragma unroll
  for (int v = 0; v < 8; ++v) {
    const int k = ((v >> 2) << 4) | (hi << 3) | ((v & 3) << 1);
    f.u[v] = *(const unsigned*)(p + k);
  }
  return f.v;
}

__device__ inline v8f wmma_f16(v16h a, v16h b, v8f c) {
  return __builtin_amdgcn_wmma_f32_16x16x32_f16(
      false, a, false, b, (short)0, c, false, false);
}

// ---------------------------------------------------------------------------
// Tensor Data Mover: 2D tile (tile_d0 x tile_d1, 16-bit elems) from a row-major
// tensor into LDS, padding each 32-half row (16 DWORDs) with 4 DWORDs so the
// LDS row stride is 40 halves (matches the frag loaders' stride).
// D# layout per CDNA5 ISA sec.8 (group0 128b, group1 256b).
// All inputs must be wave-uniform (SGPR operands).
// ---------------------------------------------------------------------------
__device__ inline void tdm_load_2d(unsigned lds_off, const void* gptr,
                                   unsigned tensor_d0, unsigned tensor_d1,
                                   unsigned tile_d0, unsigned tile_d1,
                                   unsigned row_stride_elems) {
  const unsigned long long ga = (unsigned long long)(uintptr_t)gptr;
  su4 g0;
  g0[0] = 1u;                                      // count=1 (valid descriptor)
  g0[1] = lds_off;                                 // lds_addr [63:32]
  g0[2] = (unsigned)ga;                            // global_addr [95:64]
  g0[3] = (unsigned)((ga >> 32) & 0x01FFFFFFu)     // global_addr [120:96]
          | (2u << 30);                            // type=2 ("image")
  su8 g1;
  g1[0] = (1u << 16)                               // data_size = 2 bytes
        | (1u << 20)                               // pad_enable
        | (3u << 22)                               // pad_interval: 16 DWORDs
        | (3u << 25);                              // pad_amount:    4 DWORDs
  g1[1] = (tensor_d0 & 0xFFFFu) << 16;             // tensor_dim0 [79:48] lo
  g1[2] = (tensor_d0 >> 16) | ((tensor_d1 & 0xFFFFu) << 16);  // dim0 hi|dim1 lo
  g1[3] = (tensor_d1 >> 16) | (tile_d0 << 16);     // dim1 hi | tile_dim0
  g1[4] = tile_d1;                                 // tile_dim1 (tile_dim2=0)
  g1[5] = row_stride_elems;                        // tensor_dim0_stride lo32
  g1[6] = 0u;
  g1[7] = 0u;
  asm volatile("tensor_load_to_lds %0, %1" :: "s"(g0), "s"(g1) : "memory");
}

// Per-lane async DMA of 16 bytes global -> LDS (ASYNCcnt).
__device__ inline void async_copy_b128(void* lds_dst, const void* gsrc) {
  const unsigned lds = (unsigned)(uintptr_t)lds_dst;  // low 32 = LDS offset
  const unsigned long long ga = (unsigned long long)(uintptr_t)gsrc;
  asm volatile("global_load_async_to_lds_b128 %0, %1, off"
               :: "v"(lds), "v"(ga) : "memory");
}

// ---------------------------------------------------------------------------
__global__ void cast_f32_to_f16(const float* __restrict__ src,
                                _Float16* __restrict__ dst, int n) {
  int i = blockIdx.x * blockDim.x + threadIdx.x;
  const int stride = gridDim.x * blockDim.x;
  for (; i < n; i += stride) dst[i] = (_Float16)src[i];
}

// ---------------------------------------------------------------------------
// C[M,N] = A[M,K] * W[N,K]^T   (torch nn.Linear convention).
// Block tile 128x128, 8 waves, wave tile 64x32 (4x2 WMMA accums).
// Tiles staged by TDM (wave 0), double-buffered: TDM for tile kt+1 overlaps
// WMMA on tile kt; TENSORcnt + workgroup barrier provide the handoff.
// ---------------------------------------------------------------------------
template <bool OUT_F32>
__global__ __launch_bounds__(256) void gemm_xwT(
    const _Float16* __restrict__ A, const _Float16* __restrict__ W,
    void* __restrict__ Cout, int M, int N, int K) {
  __shared__ _Float16 At[2][128][40];  // [buf][m][k], 40-halves row stride
  __shared__ _Float16 Bt[2][128][40];  // [buf][n][k]

  const int m0   = blockIdx.x * 128;
  const int n0   = blockIdx.y * 128;
  const int wave = threadIdx.x >> 5;
  const int lane = threadIdx.x & 31;
  const int hi   = lane >> 4;
  const int lo   = lane & 15;
  const int wm   = wave & 1;   // 0..1 -> 64 rows each
  const int wn   = wave >> 1;  // 0..3 -> 32 cols each

  const _Float16* Abase = A + (size_t)m0 * K;
  const _Float16* Wbase = W + (size_t)n0 * K;

  v8f acc[4][2] = {};

  const int ksteps = K >> 5;
  if (wave == 0) {  // prologue: DMA tile 0 into buffer 0
    tdm_load_2d((unsigned)(uintptr_t)&At[0][0][0], Abase, K, M, 32, 128, K);
    tdm_load_2d((unsigned)(uintptr_t)&Bt[0][0][0], Wbase, K, N, 32, 128, K);
  }

  for (int kt = 0; kt < ksteps; ++kt) {
    const int buf = kt & 1;
    if (wave == 0) __builtin_amdgcn_s_wait_tensorcnt(0);  // tile kt resident
    __syncthreads();  // also fences: every wave done reading buf^1 (tile kt-1)
    if (wave == 0 && kt + 1 < ksteps) {  // DMA tile kt+1 into other buffer
      const int k0n = (kt + 1) << 5;
      tdm_load_2d((unsigned)(uintptr_t)&At[buf ^ 1][0][0], Abase + k0n,
                  K, M, 32, 128, K);
      tdm_load_2d((unsigned)(uintptr_t)&Bt[buf ^ 1][0][0], Wbase + k0n,
                  K, N, 32, 128, K);
    }

    v16h af[4], bf[2];
#pragma unroll
    for (int i = 0; i < 4; ++i)
      af[i] = load_fragA(&At[buf][wm * 64 + i * 16][0], 40);
#pragma unroll
    for (int j = 0; j < 2; ++j)
      bf[j] = load_fragB(&Bt[buf][wn * 32 + j * 16][0], 40);
#pragma unroll
    for (int i = 0; i < 4; ++i)
#pragma unroll
      for (int j = 0; j < 2; ++j) acc[i][j] = wmma_f16(af[i], bf[j], acc[i][j]);
  }

  // C/D layout: vgpr r -> row r (lanes 0-15) or r+8 (lanes 16-31); col = lane&15.
#pragma unroll
  for (int i = 0; i < 4; ++i)
#pragma unroll
    for (int j = 0; j < 2; ++j)
#pragma unroll
      for (int r = 0; r < 8; ++r) {
        const int m = m0 + wm * 64 + i * 16 + r + hi * 8;
        const int n = n0 + wn * 32 + j * 16 + lo;
        if (OUT_F32)
          ((float*)Cout)[(size_t)m * N + n] = acc[i][j][r];
        else
          ((_Float16*)Cout)[(size_t)m * N + n] = (_Float16)acc[i][j][r];
      }
}

// ---------------------------------------------------------------------------
// Flash attention. Block = one (b, h, 64-row q tile); 4 waves x 16 q rows.
// qkv: [B*S, 3072] f16, cols 0..1023 = Q (h*64+d), 1024.. = K, 2048.. = V.
// K tile staged via per-lane async global->LDS DMA (ASYNCcnt); V transposed
// manually (DMA engines can't transpose into the [d][kv] layout we need).
// ---------------------------------------------------------------------------
__global__ __launch_bounds__(128) void attn_kernel(
    const _Float16* __restrict__ qkv, _Float16* __restrict__ out) {
  constexpr int S = 2048, DH = 64, TD = 3072, D = 1024;
  __shared__ _Float16 Kt[64][72];      // [kv][d]   (B-frag for Q*K^T)
  __shared__ _Float16 Vt[64][72];      // [d][kv]   transposed (B-frag for P*V)
  __shared__ _Float16 Pt[4][16][72];   // per-wave probs, [q][kv]

  const int bh   = blockIdx.y;
  const int b    = bh >> 4;
  const int h    = bh & 15;
  const int q0   = blockIdx.x * 64;
  const int wave = threadIdx.x >> 5;
  const int lane = threadIdx.x & 31;
  const int hi   = lane >> 4;
  const int lo   = lane & 15;

  // Q fragments for this wave's 16 rows: A-matrix 16x32, two K-steps (d 0..63).
  const _Float16* qbase = qkv + (size_t)(b * S + q0 + wave * 16) * TD + h * DH;
  v16h aq[2];
  aq[0] = load_fragA(qbase, TD);
  aq[1] = load_fragA(qbase + 32, TD);

  v8f   O[4] = {};
  float Mr[8], Lr[8];
#pragma unroll
  for (int r = 0; r < 8; ++r) { Mr[r] = -1e30f; Lr[r] = 0.f; }

  const int nkv = blockIdx.x + 1;  // only tiles with kv0 <= q0 (causal)
  for (int kt = 0; kt < nkv; ++kt) {
    const int kv0 = kt * 64;
    __syncthreads();
    // Stage 64x64 K tile (async DMA) and V tile (load + transpose store).
#pragma unroll
    for (int i = 0; i < 4; ++i) {
      const int idx = threadIdx.x + i * 128;  // 0..511 vec8 slots
      const int r = idx >> 3;                 // kv row 0..63
      const int c = (idx & 7) << 3;           // d col
      const _Float16* kp =
          qkv + (size_t)(b * S + kv0 + r) * TD + D + h * DH + c;
      async_copy_b128(&Kt[r][c], kp);         // K: global -> LDS, ASYNCcnt
      const h8 vv = *(const h8*)(kp + D);     // V (at +2048 from Q base)
#pragma unroll
      for (int j = 0; j < 8; ++j) Vt[c + j][r] = vv[j];
      __builtin_prefetch(kp + (size_t)64 * TD, 0, 1);  // next kv tile
    }
    asm volatile("s_wait_asynccnt 0x0" ::: "memory");
    __syncthreads();

    // scores[16 x 64] for this wave = Q(16x64) * K^T, scaled + causal mask.
    float s[4][8];
#pragma unroll
    for (int nt = 0; nt < 4; ++nt) {
      v8f sc = {};
      sc = wmma_f16(aq[0], load_fragB(&Kt[nt * 16][0], 72), sc);
      sc = wmma_f16(aq[1], load_fragB(&Kt[nt * 16][32], 72), sc);
      const int kcol = kv0 + nt * 16 + lo;
#pragma unroll
      for (int r = 0; r < 8; ++r) {
        const int q = q0 + wave * 16 + r + hi * 8;
        s[nt][r] = (kcol > q) ? -1e30f : sc[r] * 0.125f;  // 1/sqrt(64)
      }
    }

    // Online softmax. A C-matrix row lives across one 16-lane half.
#pragma unroll
    for (int r = 0; r < 8; ++r) {
      float mx = fmaxf(fmaxf(s[0][r], s[1][r]), fmaxf(s[2][r], s[3][r]));
#pragma unroll
      for (int d = 1; d < 16; d <<= 1) mx = fmaxf(mx, __shfl_xor(mx, d, 16));
      const float mnew  = fmaxf(Mr[r], mx);
      const float scale = __expf(Mr[r] - mnew);
      Mr[r] = mnew;
      float rs = 0.f;
#pragma unroll
      for (int nt = 0; nt < 4; ++nt) {
        const float p = __expf(s[nt][r] - mnew);
        s[nt][r] = p;
        rs += p;
      }
#pragma unroll
      for (int d = 1; d < 16; d <<= 1) rs += __shfl_xor(rs, d, 16);
      Lr[r] = Lr[r] * scale + rs;
#pragma unroll
      for (int dt = 0; dt < 4; ++dt) O[dt][r] *= scale;
    }

    // Bounce P through LDS to convert C-layout -> A-layout (per-wave region;
    // LDS ops from one wave are in-order, no barrier needed).
#pragma unroll
    for (int nt = 0; nt < 4; ++nt)
#pragma unroll
      for (int r = 0; r < 8; ++r)
        Pt[wave][r + hi * 8][nt * 16 + lo] = (_Float16)s[nt][r];

    // O += P(16x64) * V(64x64), two K-steps of 32.
#pragma unroll
    for (int ks = 0; ks < 2; ++ks) {
      const v16h pa = load_fragA(&Pt[wave][0][ks * 32], 72);
#pragma unroll
      for (int dt = 0; dt < 4; ++dt)
        O[dt] = wmma_f16(pa, load_fragB(&Vt[dt * 16][ks * 32], 72), O[dt]);
    }
  }

  // Normalize and write [b,s,h*64+d] f16.
#pragma unroll
  for (int r = 0; r < 8; ++r) {
    const float inv = 1.0f / Lr[r];
    const int q = q0 + wave * 16 + r + hi * 8;
#pragma unroll
    for (int dt = 0; dt < 4; ++dt) {
      const int d = dt * 16 + lo;
      out[(size_t)(b * S + q) * D + h * DH + d] = (_Float16)(O[dt][r] * inv);
    }
  }
}

// ---------------------------------------------------------------------------
extern "C" void kernel_launch(void* const* d_in, const int* in_sizes, int n_in,
                              void* d_out, int out_size, void* d_ws,
                              size_t ws_size, hipStream_t stream) {
  (void)in_sizes; (void)n_in; (void)out_size; (void)ws_size;
  const float* x    = (const float*)d_in[0];  // [4,2048,1024]
  const float* Wqkv = (const float*)d_in[1];  // [3072,1024]
  const float* Wo   = (const float*)d_in[2];  // [1024,1024]

  constexpr size_t MS = 8192;  // B*S
  constexpr size_t Dm = 1024, TD = 3072;

  char* ws = (char*)d_ws;
  _Float16* xh    = (_Float16*)ws; ws += MS * Dm * 2;   // 16 MB
  _Float16* wqkvh = (_Float16*)ws; ws += TD * Dm * 2;   //  6 MB
  _Float16* woh   = (_Float16*)ws; ws += Dm * Dm * 2;   //  2 MB
  _Float16* qkvh  = (_Float16*)ws; ws += MS * TD * 2;   // 48 MB
  _Float16* atth  = (_Float16*)ws;                      // 16 MB

  cast_f32_to_f16<<<512, 256, 0, stream>>>(x,    xh,    (int)(MS * Dm));
  cast_f32_to_f16<<<512, 256, 0, stream>>>(Wqkv, wqkvh, (int)(TD * Dm));
  cast_f32_to_f16<<<256, 256, 0, stream>>>(Wo,   woh,   (int)(Dm * Dm));

  // QKV projection: [8192,1024] x [3072,1024]^T -> f16 [8192,3072]
  gemm_xwT<false><<<dim3(64, 24), 256, 0, stream>>>(xh, wqkvh, (void*)qkvh,
                                                    8192, 3072, 1024);
  // Flash attention: 32 q-tiles x 64 (b,h) blocks
  attn_kernel<<<dim3(32, 64), 128, 0, stream>>>(qkvh, atth);
  // Output projection: [8192,1024] x [1024,1024]^T -> f32 d_out
  gemm_xwT<true><<<dim3(64, 8), 256, 0, stream>>>(atth, woh, d_out,
                                                  8192, 1024, 1024);
}
